// LSTMRecurrentModel_80539226735445
// MI455X (gfx1250) — matrix operand order repaired and checked by reading, not compiled, not run on hardware
//
#include <hip/hip_runtime.h>
#include <hip/hip_bf16.h>

// ---------------------------------------------------------------------------
// LSTM scan, persistent-kernel formulation for MI455X (gfx1250, wave32).
//   L=1024 timesteps, N=64 batch, D=1024 hidden, gates = 4D = 4096.
//   y_t = x_t @ Wi + h_{t-1} @ Wh + b   (bf16 WMMA, f32 accumulate)
// Data path per step:
//   gate phase (t-1) writes h and x_t as bf16 in WMMA A-operand swizzle ->
//   TDM copies the 128 KB A image into each WG's LDS once ->
//   512 waves run 128 WMMAs each from LDS(A) + packed-global(B).
// ---------------------------------------------------------------------------

typedef __bf16 bf16;
typedef __attribute__((ext_vector_type(16))) __bf16 v16bf;
typedef __attribute__((ext_vector_type(8)))  float  v8f;
typedef __attribute__((ext_vector_type(4)))  unsigned int u32x4;
typedef __attribute__((ext_vector_type(8)))  int  i32x8;
typedef __attribute__((ext_vector_type(4)))  int  i32x4;

#define L_SEQ  1024
#define NB     64          // batch rows
#define DM     1024        // d_model / hidden
#define FD     4096        // 4*D (gate columns)
#define KTILES 32          // DM / 32  (K per WMMA = 32)
#define NTILES 256         // FD / 16
#define NBLK   64          // persistent workgroups (all co-resident)
#define TPB    256         // 8 waves per workgroup

#define A_BYTES   131072u                // 64 x 1024 bf16, swizzled (128 KB)
#define A_ELEMS   65536u                 // bf16 elements

// ---- workspace layout (bytes) ---------------------------------------------
#define WS_WIP   (0u)                              // 8 MB packed bf16 Wi (B layout)
#define WS_WHP   (8u  << 20)                       // 8 MB packed bf16 Wh
#define WS_XS0   (16u << 20)                       // 128 KB swizzled bf16 x_t (ping)
#define WS_XS1   ((16u << 20) + A_BYTES)           // 128 KB swizzled bf16 x_t (pong)
#define WS_HS    ((16u << 20) + 2u * A_BYTES)      // 128 KB swizzled bf16 h
#define WS_Y0    ((16u << 20) + 3u * A_BYTES)      // 1 MB f32 partial y (x*Wi)
#define WS_Y1    (WS_Y0 + (1u << 20))              // 1 MB f32 partial y (h*Wh)
#define WS_BAR   (WS_Y1 + (1u << 20))              // barrier state

// ---------------------------------------------------------------------------
// Device-wide sense barrier (64 blocks, agent scope).
// ---------------------------------------------------------------------------
__device__ __forceinline__ void grid_barrier(unsigned* bar) {
  __threadfence();
  __syncthreads();
  if (threadIdx.x == 0) {
    unsigned* cnt = bar;
    unsigned* gen = bar + 1;
    unsigned g = __hip_atomic_load(gen, __ATOMIC_ACQUIRE, __HIP_MEMORY_SCOPE_AGENT);
    unsigned a = __hip_atomic_fetch_add(cnt, 1u, __ATOMIC_ACQ_REL, __HIP_MEMORY_SCOPE_AGENT);
    if (a == NBLK - 1) {
      __hip_atomic_store(cnt, 0u, __ATOMIC_RELAXED, __HIP_MEMORY_SCOPE_AGENT);
      __hip_atomic_fetch_add(gen, 1u, __ATOMIC_ACQ_REL, __HIP_MEMORY_SCOPE_AGENT);
    } else {
      while (__hip_atomic_load(gen, __ATOMIC_ACQUIRE, __HIP_MEMORY_SCOPE_AGENT) == g) {
        __builtin_amdgcn_s_sleep(2);
      }
    }
  }
  __syncthreads();
}

// ---------------------------------------------------------------------------
// A-operand swizzle: element (m,k) of the 64xDM matrix -> bf16 offset in the
// tiled A image. CDNA5 16-bit A 16x32 layout: lanes 0-15: M=lane, elems
// [0..7]=K0..7,[8..15]=K16..23; lanes 16-31: M=lane-16, K8..15 / K24..31.
// Per (mt,kt) tile: 32 lanes x 16 bf16 (32 B per lane, contiguous).
// ---------------------------------------------------------------------------
__device__ __forceinline__ unsigned a_swizzle(int m, int k) {
  const int mt = m >> 4, kt = k >> 5;
  const int mm = m & 15, kk = k & 31;
  const int lane = mm + (((kk >> 3) & 1) << 4);
  const int e    = (kk & 7) + ((kk >> 4) << 3);
  return (unsigned)((((mt * KTILES + kt) << 5) + lane) << 4) + (unsigned)e;
}

__device__ __forceinline__ float sigmoid_f(float x) {
  return 1.0f / (1.0f + __expf(-x));
}
__device__ __forceinline__ float tanh_f(float x) {
  return 1.0f - 2.0f / (__expf(2.0f * x) + 1.0f);
}

// ---------------------------------------------------------------------------
// TDM: DMA a contiguous block (bytes, multiple of 8) global -> LDS.
// D# group0: count=1, lds_addr, global_addr, type=2. Group1: data_size=8B,
// 1-D tensor: tensor_dim0 = tile_dim0 = bytes/8 units. Groups 2/3 unused
// (<=2-D tensor). Toolchain builtin takes 6 args (clang-23 / therock-10.0).
// ---------------------------------------------------------------------------
#if __has_builtin(__builtin_amdgcn_tensor_load_to_lds)
#define HAVE_TDM 1
__device__ __forceinline__ void tdm_load_to_lds(unsigned lds_off,
                                                const void* gsrc,
                                                unsigned bytes) {
  const unsigned long long ga = (unsigned long long)(uintptr_t)gsrc;
  const unsigned units = bytes >> 3;             // 8-byte elements
  u32x4 g0;
  g0[0] = 1u;                                    // count=1, user descriptor
  g0[1] = lds_off;                               // lds_addr
  g0[2] = (unsigned)ga;                          // global_addr[31:0]
  g0[3] = (unsigned)((ga >> 32) & 0x01FFFFFFu) | (2u << 30);  // addr[56:32]|type=2
  i32x8 g1 = {};
  g1[0] = (int)(3u << 16);                       // data_size = 3 (8 bytes)
  g1[1] = (int)(units << 16);                    // tensor_dim0[15:0]
  g1[2] = (int)(units >> 16);                    // tensor_dim0[31:16]
  g1[3] = (int)(units << 16);                    // tile_dim0 (16-bit)
  i32x4 gz4 = {};
  i32x8 gz8 = {};
  __builtin_amdgcn_tensor_load_to_lds(g0, g1, gz4, gz4, gz8, 0);
}
#else
#define HAVE_TDM 0
#endif

// ---------------------------------------------------------------------------
// Kernel 1: pack Wi/Wh into WMMA B-operand layout (bf16), stage x_0 into the
// swizzled A image, zero h image + barrier.
// B layout per 32x16 tile (1 KB): lane l<16: col n=nt*16+l, elems [0..7] -> K
// kt*32+0..7, [8..15] -> K16..23; lane l>=16: col nt*16+(l-16), K8..15 / K24..31.
// ---------------------------------------------------------------------------
__global__ void lstm_pack_init(const float* __restrict__ x,
                               const float* __restrict__ Wi,
                               const float* __restrict__ Wh,
                               bf16* __restrict__ WiP, bf16* __restrict__ WhP,
                               bf16* __restrict__ xs0, unsigned* __restrict__ hs,
                               unsigned* __restrict__ bar) {
  const int idx = blockIdx.x * blockDim.x + threadIdx.x;   // 0 .. 524287
  {  // ---- pack: 2 weights x 8192 tiles x 32 lanes ------------------------
    const int w    = idx >> 18;
    const int rem  = idx & 0x3FFFF;
    const int tile = rem >> 5;                // kt*256 + nt
    const int lane = rem & 31;
    const int kt   = tile >> 8;
    const int nt   = tile & 255;
    const float* src = w ? Wh : Wi;
    bf16* dst = (w ? WhP : WiP) + (size_t)tile * 512 + lane * 16;
    const int n  = nt * 16 + (lane & 15);
    const int kb = kt * 32 + ((lane >> 4) << 3);
#pragma unroll
    for (int e = 0; e < 16; ++e)
      dst[e] = (__bf16)src[(size_t)(kb + (e & 7) + ((e >> 3) << 4)) * FD + n];
  }
  if (idx < (int)A_ELEMS) {                    // stage x_0 (swizzled bf16)
    const int n = idx >> 10, d = idx & 1023;
    ((bf16*)xs0)[a_swizzle(n, d)] = (__bf16)x[idx];
  }
  if (idx < (int)(A_BYTES / 4)) hs[idx] = 0u;  // zero h image
  if (idx < 2) bar[idx] = 0u;
}

// ---------------------------------------------------------------------------
// Kernel 2: persistent LSTM scan. 64 WGs x 256 thr = 512 waves.
//   wave w: n-tile = w%256 (16 gate cols), half = w/256 (0: x*Wi, 1: h*Wh).
//   Per WG: TDM-stage the 128 KB swizzled A image into LDS, then each wave
//   accumulates 4 m-tiles (all 64 batch rows) = 128 WMMAs/step from LDS.
// ---------------------------------------------------------------------------
__global__ __launch_bounds__(TPB, 1)
void lstm_scan(const float* __restrict__ x, const float* __restrict__ b,
               const bf16* __restrict__ WiP, const bf16* __restrict__ WhP,
               bf16* __restrict__ xstage,      // 2 x A image (ping/pong)
               bf16* __restrict__ hstage,      // 1 x A image
               float* __restrict__ y0, float* __restrict__ y1,
               unsigned* __restrict__ bar, float* __restrict__ out) {
  extern __shared__ bf16 lbuf[];               // 128 KB A image in LDS

  const int tid   = blockIdx.x * TPB + threadIdx.x;   // 0..16383
  const int lane  = threadIdx.x & 31;
  const int wid   = tid >> 5;                         // 0..511
  const int ntile = wid & 255;
  const int half  = wid >> 8;                         // uniform per WG
  const bf16* __restrict__ Bp = half ? WhP : WiP;
  float* __restrict__ yh      = half ? y1 : y0;
  const int col   = ntile * 16 + (lane & 15);
  const int rbase = (lane >> 4) << 3;
  const v16bf* lA = (const v16bf*)lbuf;               // 32 B per lane per tile
  const unsigned lds_off = (unsigned)(uintptr_t)lbuf;

  // ---- per-thread persistent state: cell state + bias (in registers) -----
  float creg[4];
  float bia[4], bfo[4], bgg[4], boo[4];
#pragma unroll
  for (int k = 0; k < 4; ++k) {
    creg[k] = 0.0f;
    const int d = (tid + k * (NBLK * TPB)) & 1023;
    bia[k] = b[d];
    bfo[k] = b[d + 1024];
    bgg[k] = b[d + 2048];
    boo[k] = b[d + 3072];
  }

#pragma unroll 1
  for (int t = 0; t < L_SEQ; ++t) {
    // ---------------- stage A image into LDS (TDM, once per WG) ----------
    const bf16* __restrict__ Asrc =
        half ? hstage : (xstage + (size_t)(t & 1) * A_ELEMS);
#if HAVE_TDM
    if (threadIdx.x == 0) {
      tdm_load_to_lds(lds_off, Asrc, A_BYTES);
      __builtin_amdgcn_s_wait_tensorcnt(0);
    }
#else
    for (unsigned i = threadIdx.x; i < A_BYTES / 16; i += TPB)
      ((int4*)lbuf)[i] = ((const int4*)Asrc)[i];
#endif
    __syncthreads();

    // ---------------- GEMM phase: partial y = A(64xDM) * B(DMx16) --------
    v8f acc0 = {}, acc1 = {}, acc2 = {}, acc3 = {};
#pragma unroll 2
    for (int kt = 0; kt < KTILES; ++kt) {
      const v16bf bt =
          *(const v16bf*)(Bp + ((size_t)(kt * NTILES + ntile)) * 512 + lane * 16);
      const v16bf a0 = lA[((0 * KTILES + kt) << 5) + lane];
      const v16bf a1 = lA[((1 * KTILES + kt) << 5) + lane];
      const v16bf a2 = lA[((2 * KTILES + kt) << 5) + lane];
      const v16bf a3 = lA[((3 * KTILES + kt) << 5) + lane];
      acc0 = __builtin_amdgcn_wmma_f32_16x16x32_bf16(false, a0, false, bt,
                                                     (short)0, acc0, false, false);
      acc1 = __builtin_amdgcn_wmma_f32_16x16x32_bf16(false, a1, false, bt,
                                                     (short)0, acc1, false, false);
      acc2 = __builtin_amdgcn_wmma_f32_16x16x32_bf16(false, a2, false, bt,
                                                     (short)0, acc2, false, false);
      acc3 = __builtin_amdgcn_wmma_f32_16x16x32_bf16(false, a3, false, bt,
                                                     (short)0, acc3, false, false);
    }
    // C/D layout: VGPR r -> row (lanes 0-15: r, lanes 16-31: 8+r), col=lane%16
#pragma unroll
    for (int r = 0; r < 8; ++r) {
      yh[(size_t)( 0 + rbase + r) * FD + col] = acc0[r];
      yh[(size_t)(16 + rbase + r) * FD + col] = acc1[r];
      yh[(size_t)(32 + rbase + r) * FD + col] = acc2[r];
      yh[(size_t)(48 + rbase + r) * FD + col] = acc3[r];
    }

    grid_barrier(bar);   // y complete; previous A image fully consumed

    // ---------------- gate phase: 65536 cells over 16384 threads ---------
    bf16* __restrict__ xs_next = xstage + (size_t)((t + 1) & 1) * A_ELEMS;
#pragma unroll
    for (int k = 0; k < 4; ++k) {
      const int e = tid + k * (NBLK * TPB);   // n*1024 + d
      const int n = e >> 10;
      const int d = e & 1023;
      const size_t base = (size_t)n * FD + d;
      float gi = y0[base]        + y1[base]        + bia[k];
      float gf = y0[base + 1024] + y1[base + 1024] + bfo[k];
      float gg = y0[base + 2048] + y1[base + 2048] + bgg[k];
      float go = y0[base + 3072] + y1[base + 3072] + boo[k];
      gi = sigmoid_f(gi);
      gf = sigmoid_f(gf);
      gg = tanh_f(gg);
      go = sigmoid_f(go);
      const float c = gf * creg[k] + gi * gg;
      const float h = go * tanh_f(c);
      creg[k] = c;
      const unsigned sw = a_swizzle(n, d);
      hstage[sw] = (__bf16)h;                          // bf16 A image for t+1
      if (t + 1 < L_SEQ)                               // stage x_{t+1}
        xs_next[sw] = (__bf16)x[(size_t)(t + 1) * (NB * DM) + e];
      out[(size_t)t * (NB * DM) + e] = h;
    }

    grid_barrier(bar);   // h / x images visible before next step's TDM
  }
}

// ---------------------------------------------------------------------------
extern "C" void kernel_launch(void* const* d_in, const int* in_sizes, int n_in,
                              void* d_out, int out_size, void* d_ws, size_t ws_size,
                              hipStream_t stream) {
  const float* x  = (const float*)d_in[0];   // [L, N, D]
  const float* Wi = (const float*)d_in[1];   // [D, 4D]
  const float* Wh = (const float*)d_in[2];   // [D, 4D]
  const float* b  = (const float*)d_in[3];   // [4D]
  float* out = (float*)d_out;                // [L, N, D]

  char* ws = (char*)d_ws;
  bf16*     WiP  = (bf16*)(ws + WS_WIP);
  bf16*     WhP  = (bf16*)(ws + WS_WHP);
  bf16*     xs   = (bf16*)(ws + WS_XS0);     // ping/pong pair (contiguous)
  bf16*     hs   = (bf16*)(ws + WS_HS);
  float*    y0   = (float*)(ws + WS_Y0);
  float*    y1   = (float*)(ws + WS_Y1);
  unsigned* bar  = (unsigned*)(ws + WS_BAR);

  static int lds_cfg = 0;
  if (!lds_cfg) {   // idempotent attribute set (host-side, not a stream op)
    (void)hipFuncSetAttribute((const void*)lstm_scan,
                              hipFuncAttributeMaxDynamicSharedMemorySize,
                              (int)A_BYTES);
    lds_cfg = 1;
  }

  lstm_pack_init<<<2048, 256, 0, stream>>>(x, Wi, Wh, WiP, WhP, xs, (unsigned*)hs, bar);
  lstm_scan<<<NBLK, TPB, A_BYTES, stream>>>(x, b, WiP, WhP, xs, hs, y0, y1, bar, out);
}
